// LiquidityGraphOperator_4535485465148
// MI455X (gfx1250) — matrix operand order repaired and checked by reading, not compiled, not run on hardware
//
#include <hip/hip_runtime.h>
#include <hip/hip_bf16.h>
#include <math.h>

// ---------------- constants (match reference) ----------------
#define NND   40000
#define EED   320000
#define NTD   3
#define LD    3
#define DD    128
#define HIDD  256
#define HD    4
#define CD    64

typedef __attribute__((ext_vector_type(16))) __bf16 v16bf;
typedef __attribute__((ext_vector_type(8)))  float  v8f;

union BFrag { v16bf v; unsigned int u[8]; };

__device__ __forceinline__ float gelu_exact(float x) {
  return 0.5f * x * (1.0f + erff(x * 0.70710678118654752f));
}

__device__ __forceinline__ unsigned short f2bf(float f) {
  unsigned int u = __float_as_uint(f);
  unsigned int r = 0x7FFFu + ((u >> 16) & 1u);   // round-to-nearest-even
  return (unsigned short)((u + r) >> 16);
}

__device__ __forceinline__ void atomicMaxF(float* addr, float val) {
  // order-preserving int trick (works with -inf init)
  if (val >= 0.0f) atomicMax((int*)addr, __float_as_int(val));
  else             atomicMin((unsigned int*)addr, __float_as_uint(val));
}

// all-thread block reduction; blockDim.x == 256
__device__ __forceinline__ float block_reduce_sum(float v, float* red) {
  int c = threadIdx.x;
  red[c] = v; __syncthreads();
  for (int s = 128; s > 0; s >>= 1) { if (c < s) red[c] += red[c + s]; __syncthreads(); }
  float r = red[0]; __syncthreads();
  return r;
}

// ---------------- utility kernels ----------------
__global__ void k_fill(float* __restrict__ p, float v, int n) {
  int t = blockIdx.x * blockDim.x + threadIdx.x;
  if (t < n) p[t] = v;
}

__global__ void k_f32bf(const float* __restrict__ s, unsigned short* __restrict__ d, int n) {
  int t = blockIdx.x * blockDim.x + threadIdx.x;
  if (t < n) d[t] = f2bf(s[t]);
}

// W[K x Nout] row-major -> Wt[Nout x K] bf16 (so GEMM B loads are K-contiguous)
__global__ void k_convtr(const float* __restrict__ W, unsigned short* __restrict__ Wt,
                         int K, int Nout) {
  int t = blockIdx.x * blockDim.x + threadIdx.x;
  if (t >= K * Nout) return;
  int n = t / K, k = t % K;
  Wt[t] = f2bf(W[(size_t)k * Nout + n]);
}

// ---------------- WMMA bf16 GEMM ----------------
// C[M x Nout] = A[M x K](bf16,row-major) * Bt[Nout x K](bf16)^T + bias
// One wave computes a 32x64 macro-tile: 2 M-subtiles x 4 N-subtiles of 16x16,
// so each K-step issues 8 v_wmma per (2 A + 4 B) fragment loads.
__global__ void k_gemm_bf16(const unsigned short* __restrict__ A,
                            const unsigned short* __restrict__ Bt,
                            const float* __restrict__ bias,
                            float* __restrict__ C,
                            int M, int K, int Nout) {
  const int wave = threadIdx.x >> 5;
  const int lane = threadIdx.x & 31;
  const int tilesN = Nout >> 6;                 // 64-wide macro-tiles
  const int totalTiles = (M >> 5) * tilesN;     // 32-tall macro-tiles
  int tile = blockIdx.x * (blockDim.x >> 5) + wave;
  if (tile >= totalTiles) return;               // wave-uniform branch; EXEC stays all-ones
  const int mt = tile / tilesN;
  const int nt = tile % tilesN;
  const int hgrp = lane >> 4;                   // lane half-group
  const int l16  = lane & 15;

  const unsigned short* arow0 = A + (size_t)(mt * 32 + l16) * K;
  const unsigned short* arow1 = arow0 + (size_t)16 * K;
  const unsigned short* brow0 = Bt + (size_t)(nt * 64 + 0 * 16 + l16) * K;
  const unsigned short* brow1 = Bt + (size_t)(nt * 64 + 1 * 16 + l16) * K;
  const unsigned short* brow2 = Bt + (size_t)(nt * 64 + 2 * 16 + l16) * K;
  const unsigned short* brow3 = Bt + (size_t)(nt * 64 + 3 * 16 + l16) * K;

  v8f acc00 = {}, acc01 = {}, acc02 = {}, acc03 = {};
  v8f acc10 = {}, acc11 = {}, acc12 = {}, acc13 = {};

  for (int k0 = 0; k0 < K; k0 += 32) {
    BFrag a0, a1, b0, b1, b2, b3;
#pragma unroll
    for (int v = 0; v < 8; ++v) {
      // A 16x32 bf16 layout: VGPR v<4 -> K = hgrp*8 + 2v(+1); v>=4 -> 16 + hgrp*8 + 2(v-4)(+1)
      int ka = ((v < 4) ? 0 : 16) + hgrp * 8 + 2 * (v & 3);
      a0.u[v] = *(const unsigned int*)(arow0 + k0 + ka);
      a1.u[v] = *(const unsigned int*)(arow1 + k0 + ka);
      // B 32x16 bf16 layout (same as A): VGPR v<4 -> K = hgrp*8 + 2v(+1); v>=4 -> 16 + hgrp*8 + 2(v-4)(+1)
      int kb = ((v < 4) ? 0 : 16) + hgrp * 8 + 2 * (v & 3);
      b0.u[v] = *(const unsigned int*)(brow0 + k0 + kb);
      b1.u[v] = *(const unsigned int*)(brow1 + k0 + kb);
      b2.u[v] = *(const unsigned int*)(brow2 + k0 + kb);
      b3.u[v] = *(const unsigned int*)(brow3 + k0 + kb);
    }
    acc00 = __builtin_amdgcn_wmma_f32_16x16x32_bf16(false, a0.v, false, b0.v, (short)0, acc00, false, false);
    acc01 = __builtin_amdgcn_wmma_f32_16x16x32_bf16(false, a0.v, false, b1.v, (short)0, acc01, false, false);
    acc02 = __builtin_amdgcn_wmma_f32_16x16x32_bf16(false, a0.v, false, b2.v, (short)0, acc02, false, false);
    acc03 = __builtin_amdgcn_wmma_f32_16x16x32_bf16(false, a0.v, false, b3.v, (short)0, acc03, false, false);
    acc10 = __builtin_amdgcn_wmma_f32_16x16x32_bf16(false, a1.v, false, b0.v, (short)0, acc10, false, false);
    acc11 = __builtin_amdgcn_wmma_f32_16x16x32_bf16(false, a1.v, false, b1.v, (short)0, acc11, false, false);
    acc12 = __builtin_amdgcn_wmma_f32_16x16x32_bf16(false, a1.v, false, b2.v, (short)0, acc12, false, false);
    acc13 = __builtin_amdgcn_wmma_f32_16x16x32_bf16(false, a1.v, false, b3.v, (short)0, acc13, false, false);
  }

  // epilogue: C layout per 16x16 f32 tile: VGPR r -> row r (lanes 0-15) / 8+r (lanes 16-31)
  const v8f* accs[2][4] = {{&acc00, &acc01, &acc02, &acc03},
                           {&acc10, &acc11, &acc12, &acc13}};
#pragma unroll
  for (int mi = 0; mi < 2; ++mi) {
#pragma unroll
    for (int j = 0; j < 4; ++j) {
      const int col = nt * 64 + j * 16 + l16;
      const float bv = bias ? bias[col] : 0.0f;
      const v8f acc = *accs[mi][j];
#pragma unroll
      for (int r = 0; r < 8; ++r) {
        int row = mt * 32 + mi * 16 + hgrp * 8 + r;
        C[(size_t)row * Nout + col] = acc[r] + bv;
      }
    }
  }
}

// ---------------- row-wise LayerNorm + exact GELU (width 256) ----------------
__global__ void k_ln_gelu(const float* __restrict__ in, float* __restrict__ outp,
                          const float* __restrict__ g, const float* __restrict__ b) {
  __shared__ float red[256];
  int n = blockIdx.x, c = threadIdx.x;
  float y   = in[(size_t)n * HIDD + c];
  float m   = block_reduce_sum(y, red) * (1.0f / HIDD);
  float d   = y - m;
  float var = block_reduce_sum(d * d, red) * (1.0f / HIDD);
  float z   = g[c] * d * rsqrtf(var + 1e-5f) + b[c];
  outp[(size_t)n * HIDD + c] = gelu_exact(z);
}

// ---------------- edge-weight gate ----------------
__global__ void k_ewg_accum(const int* __restrict__ ei, const float* __restrict__ ew,
                            float* __restrict__ wsum, float* __restrict__ deg) {
  int t = blockIdx.x * blockDim.x + threadIdx.x;
  if (t >= NTD * EED) return;
  int ty = t / EED, e = t % EED;
  int d = ei[((size_t)ty * 2 + 1) * EED + e];
  atomicAdd(&wsum[d], ew[(size_t)ty * EED + e]);
  atomicAdd(&deg[d], 1.0f);
}

__global__ void k_ewg_final(const float* __restrict__ wsum, const float* __restrict__ deg,
                            float* __restrict__ ewg) {
  int t = blockIdx.x * blockDim.x + threadIdx.x;
  if (t >= NND) return;
  float z = wsum[t] / fmaxf(deg[t], 1.0f);
  ewg[t] = 0.5f + 1.0f / (1.0f + expf(-z));
}

// ---------------- GAT attention pieces ----------------
__global__ void k_scores(const float* __restrict__ xp,
                         const float* __restrict__ asrc, const float* __restrict__ adst,
                         float* __restrict__ ssrc, float* __restrict__ sdst) {
  int t = blockIdx.x * blockDim.x + threadIdx.x;
  if (t >= NND * HD) return;
  int n = t >> 2, hh = t & 3;
  const float* row = xp + (size_t)n * HIDD + hh * CD;
  const float* as  = asrc + hh * CD;
  const float* ad  = adst + hh * CD;
  float s0 = 0.0f, s1 = 0.0f;
  for (int c = 0; c < CD; ++c) { float v = row[c]; s0 += v * as[c]; s1 += v * ad[c]; }
  ssrc[t] = s0; sdst[t] = s1;
}

__device__ __forceinline__ float edge_score(const int* src, const int* dst,
                                            const float* ssrc, const float* sdst,
                                            int e, int hh, int& d) {
  int s = src[e]; d = dst[e];
  float v = ssrc[s * HD + hh] + sdst[d * HD + hh];
  return v > 0.0f ? v : 0.2f * v;   // leaky_relu 0.2
}

__global__ void k_edge_max(const int* __restrict__ src, const int* __restrict__ dst,
                           const float* __restrict__ ssrc, const float* __restrict__ sdst,
                           float* __restrict__ emax) {
  int t = blockIdx.x * blockDim.x + threadIdx.x;
  if (t >= EED * HD) return;
  int e = t >> 2, hh = t & 3, d;
  float v = edge_score(src, dst, ssrc, sdst, e, hh, d);
  atomicMaxF(&emax[d * HD + hh], v);
}

__global__ void k_emax_fix(float* __restrict__ emax, int n) {
  int t = blockIdx.x * blockDim.x + threadIdx.x;
  if (t >= n) return;
  float v = emax[t];
  if (!isfinite(v)) emax[t] = 0.0f;
}

__global__ void k_edge_expsum(const int* __restrict__ src, const int* __restrict__ dst,
                              const float* __restrict__ ssrc, const float* __restrict__ sdst,
                              const float* __restrict__ emax,
                              float* __restrict__ exb, float* __restrict__ den) {
  int t = blockIdx.x * blockDim.x + threadIdx.x;
  if (t >= EED * HD) return;
  int e = t >> 2, hh = t & 3, d;
  float v  = edge_score(src, dst, ssrc, sdst, e, hh, d);
  float ex = expf(v - emax[d * HD + hh]);
  exb[t] = ex;
  atomicAdd(&den[d * HD + hh], ex);
}

__global__ void k_edge_scatter(const int* __restrict__ src, const int* __restrict__ dst,
                               const float* __restrict__ exb, const float* __restrict__ den,
                               const float* __restrict__ xp, float* __restrict__ agg) {
  int t = blockIdx.x * blockDim.x + threadIdx.x;
  if (t >= EED * HIDD) return;
  int e = t >> 8, c = t & 255, hh = c >> 6;
  int s = src[e], d = dst[e];
  // prefetch the next edge's source row (speculative; gfx1250 global_prefetch_b8)
  if (e + 1 < EED) {
    int sn = src[e + 1];
    __builtin_prefetch(&xp[(size_t)sn * HIDD + c], 0, 0);
  }
  float alpha = exb[e * HD + hh] / (den[d * HD + hh] + 1e-16f);
  atomicAdd(&agg[(size_t)d * HIDD + c], alpha * xp[(size_t)s * HIDD + c]);
}

__global__ void k_agg_bias3(float* __restrict__ agg, const float* __restrict__ b0,
                            const float* __restrict__ b1, const float* __restrict__ b2) {
  int t = blockIdx.x * blockDim.x + threadIdx.x;
  if (t >= NND * HIDD) return;
  int c = t & 255;
  agg[t] = b0[c] + b1[c] + b2[c];
}

// ---------------- liquidity gate + ewg + residual + LN + GELU (per node) ----------------
__global__ void k_gate_ln(const float* __restrict__ agg, float* __restrict__ h,
                          const float* __restrict__ liq,
                          const float* __restrict__ gateW, const float* __restrict__ gateb,
                          const float* __restrict__ ewg,
                          const float* __restrict__ lng, const float* __restrict__ lnb,
                          float* __restrict__ lay) {
  __shared__ float red[256];
  int n = blockIdx.x, c = threadIdx.x;
  float a  = agg[(size_t)n * HIDD + c];
  float hv = h[(size_t)n * HIDD + c];
  float dot = block_reduce_sum(a * gateW[c], red);
  float gl  = dot + liq[n * 4 + 0] * gateW[256] + liq[n * 4 + 1] * gateW[257]
                  + liq[n * 4 + 2] * gateW[258] + liq[n * 4 + 3] * gateW[259] + gateb[0];
  float gate = 1.0f / (1.0f + expf(-gl));
  float hn = (gate * a + (1.0f - gate) * hv) * ewg[n];
  float y  = hn + hv;
  float m   = block_reduce_sum(y, red) * (1.0f / HIDD);
  float d   = y - m;
  float var = block_reduce_sum(d * d, red) * (1.0f / HIDD);
  float z   = lng[c] * d * rsqrtf(var + 1e-5f) + lnb[c];
  float o   = gelu_exact(z);
  h[(size_t)n * HIDD + c]   = o;
  lay[(size_t)n * HIDD + c] = o;
}

// ---------------- readout ----------------
__global__ void k_colmean(const float* __restrict__ ho, float* __restrict__ colmean,
                          int rowsPerBlock) {
  int c = threadIdx.x;                 // 0..127
  int r0 = blockIdx.x * rowsPerBlock;
  int r1 = r0 + rowsPerBlock; if (r1 > NND) r1 = NND;
  float s = 0.0f;
  for (int r = r0; r < r1; ++r) s += ho[(size_t)r * DD + c];
  atomicAdd(&colmean[c], s);
}

__global__ void k_readout(const float* __restrict__ colmean,
                          const float* __restrict__ roW1, const float* __restrict__ rob1,
                          const float* __restrict__ roW2, const float* __restrict__ rob2,
                          float* __restrict__ surf) {
  __shared__ float cm[DD];
  __shared__ float s1[HIDD];
  int c = threadIdx.x;                 // 0..255
  if (c < DD) cm[c] = colmean[c] * (1.0f / NND);
  __syncthreads();
  float acc = 0.0f;
  for (int k = 0; k < DD; ++k) acc += cm[k] * roW1[(size_t)k * HIDD + c];
  s1[c] = gelu_exact(acc + rob1[c]);
  __syncthreads();
  float acc2 = 0.0f;
  for (int k = 0; k < HIDD; ++k) acc2 += s1[k] * roW2[(size_t)k * HIDD + c];
  surf[c] = acc2 + rob2[c];
}

// ---------------- host ----------------
#define CDIV(a, b) (((a) + (b) - 1) / (b))

static void launch_gemm(const unsigned short* A, const unsigned short* Bt, const float* bias,
                        float* C, int M, int K, int Nout, hipStream_t s) {
  int tiles = (M / 32) * (Nout / 64);          // 32x64 macro-tiles per wave
  int blocks = CDIV(tiles, 8);                 // 8 waves per 256-thread block
  k_gemm_bf16<<<blocks, 256, 0, s>>>(A, Bt, bias, C, M, K, Nout);
}

extern "C" void kernel_launch(void* const* d_in, const int* in_sizes, int n_in,
                              void* d_out, int out_size, void* d_ws, size_t ws_size,
                              hipStream_t stream) {
  (void)in_sizes; (void)n_in; (void)out_size; (void)ws_size;
  const float* x        = (const float*)d_in[0];
  const float* liq      = (const float*)d_in[1];
  const int*   ei       = (const int*)  d_in[2];
  const float* ew       = (const float*)d_in[3];
  const float* in_W     = (const float*)d_in[4];
  const float* in_b     = (const float*)d_in[5];
  const float* in_g     = (const float*)d_in[6];
  const float* in_bb    = (const float*)d_in[7];
  const float* gat_W    = (const float*)d_in[8];
  const float* gat_asrc = (const float*)d_in[9];
  const float* gat_adst = (const float*)d_in[10];
  const float* gat_bias = (const float*)d_in[11];
  const float* ln_g     = (const float*)d_in[12];
  const float* ln_b     = (const float*)d_in[13];
  const float* gate_W   = (const float*)d_in[14];
  const float* gate_b   = (const float*)d_in[15];
  const float* out_W1   = (const float*)d_in[16];
  const float* out_b1   = (const float*)d_in[17];
  const float* out_g    = (const float*)d_in[18];
  const float* out_bb   = (const float*)d_in[19];
  const float* out_W2   = (const float*)d_in[20];
  const float* out_b2   = (const float*)d_in[21];
  const float* ro_W1    = (const float*)d_in[22];
  const float* ro_b1    = (const float*)d_in[23];
  const float* ro_W2    = (const float*)d_in[24];
  const float* ro_b2    = (const float*)d_in[25];

  float* out        = (float*)d_out;
  float* out_ho     = out;                               // N*D
  float* out_surf   = out + (size_t)NND * DD;            // HID
  float* out_layers = out_surf + HIDD;                   // L*N*HID

  // workspace carve-up (byte bump allocator, 256B aligned)
  char* wp = (char*)d_ws;
  auto alloc = [&](size_t bytes) -> char* {
    char* p = wp; wp += (bytes + 255) & ~(size_t)255; return p;
  };
  float* h    = (float*)alloc((size_t)NND * HIDD * 4);
  float* xp   = (float*)alloc((size_t)NND * HIDD * 4);   // also GEMM temp pre-LN
  float* agg  = (float*)alloc((size_t)NND * HIDD * 4);   // also out-head post-LN temp
  unsigned short* hb   = (unsigned short*)alloc((size_t)NND * HIDD * 2);
  unsigned short* inWt = (unsigned short*)alloc((size_t)DD * HIDD * 2);
  unsigned short* gatWt= (unsigned short*)alloc((size_t)LD * NTD * HIDD * HIDD * 2);
  unsigned short* w1t  = (unsigned short*)alloc((size_t)HIDD * HIDD * 2);
  unsigned short* w2t  = (unsigned short*)alloc((size_t)HIDD * DD * 2);
  float* ssrc = (float*)alloc((size_t)NND * HD * 4);
  float* sdst = (float*)alloc((size_t)NND * HD * 4);
  float* emax = (float*)alloc((size_t)NND * HD * 4);
  float* den  = (float*)alloc((size_t)NND * HD * 4);
  float* exb  = (float*)alloc((size_t)EED * HD * 4);
  float* wsum = (float*)alloc((size_t)NND * 4);
  float* degb = (float*)alloc((size_t)NND * 4);
  float* ewg  = (float*)alloc((size_t)NND * 4);
  float* cmn  = (float*)alloc((size_t)DD * 4);

  // ---- weight conversion (bf16, transposed) ----
  k_convtr<<<CDIV(DD * HIDD, 256), 256, 0, stream>>>(in_W, inWt, DD, HIDD);
  for (int lt = 0; lt < LD * NTD; ++lt)
    k_convtr<<<CDIV(HIDD * HIDD, 256), 256, 0, stream>>>(
        gat_W + (size_t)lt * HIDD * HIDD, gatWt + (size_t)lt * HIDD * HIDD, HIDD, HIDD);
  k_convtr<<<CDIV(HIDD * HIDD, 256), 256, 0, stream>>>(out_W1, w1t, HIDD, HIDD);
  k_convtr<<<CDIV(HIDD * DD, 256), 256, 0, stream>>>(out_W2, w2t, HIDD, DD);

  // ---- edge-weight gate (static) ----
  k_fill<<<CDIV(NND, 256), 256, 0, stream>>>(wsum, 0.0f, NND);
  k_fill<<<CDIV(NND, 256), 256, 0, stream>>>(degb, 0.0f, NND);
  k_ewg_accum<<<CDIV(NTD * EED, 256), 256, 0, stream>>>(ei, ew, wsum, degb);
  k_ewg_final<<<CDIV(NND, 256), 256, 0, stream>>>(wsum, degb, ewg);

  // ---- input projection ----
  k_f32bf<<<CDIV(NND * DD, 256), 256, 0, stream>>>(x, hb, NND * DD);
  launch_gemm(hb, inWt, in_b, xp, NND, DD, HIDD, stream);
  k_ln_gelu<<<NND, 256, 0, stream>>>(xp, h, in_g, in_bb);

  // ---- message-passing layers ----
  for (int i = 0; i < LD; ++i) {
    k_agg_bias3<<<CDIV(NND * HIDD, 256), 256, 0, stream>>>(
        agg, gat_bias + (size_t)(i * NTD + 0) * HIDD,
             gat_bias + (size_t)(i * NTD + 1) * HIDD,
             gat_bias + (size_t)(i * NTD + 2) * HIDD);
    k_f32bf<<<CDIV(NND * HIDD, 256), 256, 0, stream>>>(h, hb, NND * HIDD);
    for (int t = 0; t < NTD; ++t) {
      const int lt = i * NTD + t;
      const int* srcp = ei + ((size_t)t * 2 + 0) * EED;
      const int* dstp = ei + ((size_t)t * 2 + 1) * EED;
      launch_gemm(hb, gatWt + (size_t)lt * HIDD * HIDD, nullptr, xp, NND, HIDD, HIDD, stream);
      k_scores<<<CDIV(NND * HD, 256), 256, 0, stream>>>(
          xp, gat_asrc + (size_t)lt * HIDD, gat_adst + (size_t)lt * HIDD, ssrc, sdst);
      k_fill<<<CDIV(NND * HD, 256), 256, 0, stream>>>(emax, -INFINITY, NND * HD);
      k_fill<<<CDIV(NND * HD, 256), 256, 0, stream>>>(den, 0.0f, NND * HD);
      k_edge_max<<<CDIV(EED * HD, 256), 256, 0, stream>>>(srcp, dstp, ssrc, sdst, emax);
      k_emax_fix<<<CDIV(NND * HD, 256), 256, 0, stream>>>(emax, NND * HD);
      k_edge_expsum<<<CDIV(EED * HD, 256), 256, 0, stream>>>(srcp, dstp, ssrc, sdst, emax, exb, den);
      k_edge_scatter<<<CDIV(EED * HIDD, 256), 256, 0, stream>>>(srcp, dstp, exb, den, xp, agg);
    }
    k_gate_ln<<<NND, 256, 0, stream>>>(agg, h, liq, gate_W, gate_b, ewg,
                                       ln_g + (size_t)i * HIDD, ln_b + (size_t)i * HIDD,
                                       out_layers + (size_t)i * NND * HIDD);
  }

  // ---- output head ----
  k_f32bf<<<CDIV(NND * HIDD, 256), 256, 0, stream>>>(h, hb, NND * HIDD);
  launch_gemm(hb, w1t, out_b1, xp, NND, HIDD, HIDD, stream);
  k_ln_gelu<<<NND, 256, 0, stream>>>(xp, agg, out_g, out_bb);
  k_f32bf<<<CDIV(NND * HIDD, 256), 256, 0, stream>>>(agg, hb, NND * HIDD);
  launch_gemm(hb, w2t, out_b2, out_ho, NND, HIDD, DD, stream);

  // ---- readout ----
  k_fill<<<1, 256, 0, stream>>>(cmn, 0.0f, DD);
  k_colmean<<<CDIV(NND, 256), DD, 0, stream>>>(out_ho, cmn, 256);
  k_readout<<<1, 256, 0, stream>>>(cmn, ro_W1, ro_b1, ro_W2, ro_b2, out_surf);
}
